// GCN_GIN_44744969290572
// MI455X (gfx1250) — matrix operand order repaired and checked by reading, not compiled, not run on hardware
//
#include <hip/hip_runtime.h>
#include <hip/hip_bf16.h>

typedef __attribute__((ext_vector_type(16))) _Float16 v16h;
typedef __attribute__((ext_vector_type(8)))  _Float16 v8h;
typedef __attribute__((ext_vector_type(4)))  _Float16 v4h;
typedef __attribute__((ext_vector_type(8)))  float    v8f;

#define N_NODES   100000
#define N_EDGES   600000
#define DFEAT     128
#define DIMH      128
#define NCLS      40
#define NCLS_PAD  48
#define BN_EPS    1e-5f
#define REGBYTES  (100000ull * 128ull * 4ull)   // 51.2 MB per activation region

// ---------------------------------------------------------------------------
// Kernel 1: h0 = x (copy), and zero the BN-stat accumulators (256 floats).
// ---------------------------------------------------------------------------
__global__ void k_init(const float* __restrict__ x, float* __restrict__ h0,
                       float* __restrict__ stats) {
    size_t t = (size_t)blockIdx.x * blockDim.x + threadIdx.x;   // one float4 each
    const float4* xi = (const float4*)x;
    float4* ho = (float4*)h0;
    if (t < (size_t)N_NODES * DFEAT / 4) ho[t] = xi[t];
    if (blockIdx.x == 0 && threadIdx.x < 256) stats[threadIdx.x] = 0.0f;
}

// ---------------------------------------------------------------------------
// Kernel 2: edge scatter-add  h0[dst] += x[src].  One thread = (edge, 4 feats).
// 51 MB feature table is L2-resident (192 MB L2), atomics land in L2.
// ---------------------------------------------------------------------------
__global__ void k_scatter(const float* __restrict__ x,
                          const long long* __restrict__ ei,
                          float* __restrict__ h0) {
    size_t t = (size_t)blockIdx.x * blockDim.x + threadIdx.x;
    if (t >= (size_t)N_EDGES * 32) return;
    int e = (int)(t >> 5);
    int f = (int)(t & 31) * 4;
    int s = (int)ei[e];
    int d = (int)ei[N_EDGES + e];
    const float4 v = *(const float4*)(x + (size_t)s * DFEAT + f);
    float* p = h0 + (size_t)d * DFEAT + f;
    atomicAdd(p + 0, v.x);
    atomicAdd(p + 1, v.y);
    atomicAdd(p + 2, v.z);
    atomicAdd(p + 3, v.w);
}

// ---------------------------------------------------------------------------
// Kernel 3: h1 = h0 @ w1^T + b1  (fp32 A converted to f16, WMMA f32 accum)
// plus fused per-column sum / sum-of-squares for BatchNorm batch statistics.
// Block = 256 thr = 8 waves; each wave: 16 rows x 128 cols, K-loop of 4.
// ---------------------------------------------------------------------------
__global__ __launch_bounds__(256) void k_gemm1(const float* __restrict__ h0,
                                               const float* __restrict__ w1,
                                               const float* __restrict__ b1,
                                               float* __restrict__ h1,
                                               float* __restrict__ gsum,
                                               float* __restrict__ gsq) {
    __shared__ __align__(32) _Float16 wlds[DIMH * DFEAT];   // 32 KB f16 weights
    __shared__ float csum[DIMH];
    __shared__ float csq[DIMH];

    int tid = threadIdx.x;
    for (int i = tid; i < DIMH * DFEAT; i += 256) wlds[i] = (_Float16)w1[i];
    if (tid < DIMH) { csum[tid] = 0.0f; csq[tid] = 0.0f; }
    __syncthreads();

    int wave = tid >> 5, lane = tid & 31;
    int half = lane >> 4, ln = lane & 15;
    int rowbase = blockIdx.x * 128 + wave * 16;

    v8f acc[8] = {};
    int r = rowbase + ln;                       // A row for this lane (both halves)

    for (int kt = 0; kt < 4; ++kt) {
        v16h a = {};
        if (r < N_NODES) {
            const float* pa = h0 + (size_t)r * DFEAT + kt * 32 + 16 * half;
            #pragma unroll
            for (int i = 0; i < 8; ++i) a[i] = (_Float16)pa[i];
            #pragma unroll
            for (int i = 0; i < 8; ++i) a[8 + i] = (_Float16)pa[8 + i];
        }
        #pragma unroll
        for (int nt = 0; nt < 8; ++nt) {
            v16h b = *(const v16h*)(wlds + (nt * 16 + ln) * DFEAT + kt * 32 + 16 * half);
            acc[nt] = __builtin_amdgcn_wmma_f32_16x16x32_f16(
                false, a, false, b, (short)0, acc[nt], false, false);
        }
    }

    #pragma unroll
    for (int nt = 0; nt < 8; ++nt) {
        int col = nt * 16 + ln;
        float bias = b1[col];
        float s = 0.0f, s2 = 0.0f;
        #pragma unroll
        for (int j = 0; j < 8; ++j) {
            int row = rowbase + j + 8 * half;
            float v = acc[nt][j] + bias;
            if (row < N_NODES) {
                h1[(size_t)row * DIMH + col] = v;
                s += v; s2 += v * v;
            }
        }
        atomicAdd(&csum[col], s);
        atomicAdd(&csq[col], s2);
    }
    __syncthreads();
    if (tid < DIMH) {
        atomicAdd(&gsum[tid], csum[tid]);
        atomicAdd(&gsq[tid],  csq[tid]);
    }
}

// ---------------------------------------------------------------------------
// Kernel 4: finalize BN -> per-column scale/shift (128 cols).
// ---------------------------------------------------------------------------
__global__ void k_bnfin(const float* __restrict__ gsum, const float* __restrict__ gsq,
                        const float* __restrict__ gamma, const float* __restrict__ beta,
                        float* __restrict__ scale, float* __restrict__ shift) {
    int c = threadIdx.x;
    if (c >= DIMH) return;
    const float inv_n = 1.0f / (float)N_NODES;
    float m = gsum[c] * inv_n;
    float v = gsq[c] * inv_n - m * m;
    float sc = gamma[c] * rsqrtf(v + BN_EPS);
    scale[c] = sc;
    shift[c] = beta[c] - m * sc;
}

// ---------------------------------------------------------------------------
// Kernel 5: h1h = f16( relu(h1 * scale + shift) )
// ---------------------------------------------------------------------------
__global__ void k_bncast(const float* __restrict__ h1,
                         const float* __restrict__ scale,
                         const float* __restrict__ shift,
                         _Float16* __restrict__ h1h) {
    size_t t = (size_t)blockIdx.x * blockDim.x + threadIdx.x;
    if (t >= (size_t)N_NODES * DIMH / 4) return;
    size_t base = t * 4;
    int col = (int)(base & (DIMH - 1));
    float4 v = *(const float4*)(h1 + base);
    v4h o;
    float a;
    a = v.x * scale[col + 0] + shift[col + 0]; o[0] = (_Float16)(a > 0.f ? a : 0.f);
    a = v.y * scale[col + 1] + shift[col + 1]; o[1] = (_Float16)(a > 0.f ? a : 0.f);
    a = v.z * scale[col + 2] + shift[col + 2]; o[2] = (_Float16)(a > 0.f ? a : 0.f);
    a = v.w * scale[col + 3] + shift[col + 3]; o[3] = (_Float16)(a > 0.f ? a : 0.f);
    *(v4h*)(h1h + base) = o;
}

// ---------------------------------------------------------------------------
// Kernel 6: h2h = f16( relu(h1h @ w2^T + b2) )   (f16 in, f16 out, f32 accum)
// ---------------------------------------------------------------------------
__global__ __launch_bounds__(256) void k_gemm2(const _Float16* __restrict__ h1h,
                                               const float* __restrict__ w2,
                                               const float* __restrict__ b2,
                                               _Float16* __restrict__ h2h) {
    __shared__ __align__(32) _Float16 wlds[DIMH * DIMH];
    int tid = threadIdx.x;
    for (int i = tid; i < DIMH * DIMH; i += 256) wlds[i] = (_Float16)w2[i];
    __syncthreads();

    int wave = tid >> 5, lane = tid & 31;
    int half = lane >> 4, ln = lane & 15;
    int rowbase = blockIdx.x * 128 + wave * 16;

    v8f acc[8] = {};
    int r = rowbase + ln;

    for (int kt = 0; kt < 4; ++kt) {
        v16h a = {};
        if (r < N_NODES) {
            const _Float16* pa = h1h + (size_t)r * DIMH + kt * 32 + 16 * half;
            v8h lo = *(const v8h*)(pa);
            v8h hi = *(const v8h*)(pa + 8);
            #pragma unroll
            for (int i = 0; i < 8; ++i) { a[i] = lo[i]; a[8 + i] = hi[i]; }
        }
        #pragma unroll
        for (int nt = 0; nt < 8; ++nt) {
            v16h b = *(const v16h*)(wlds + (nt * 16 + ln) * DIMH + kt * 32 + 16 * half);
            acc[nt] = __builtin_amdgcn_wmma_f32_16x16x32_f16(
                false, a, false, b, (short)0, acc[nt], false, false);
        }
    }

    #pragma unroll
    for (int nt = 0; nt < 8; ++nt) {
        int col = nt * 16 + ln;
        float bias = b2[col];
        #pragma unroll
        for (int j = 0; j < 8; ++j) {
            int row = rowbase + j + 8 * half;
            if (row < N_NODES) {
                float v = acc[nt][j] + bias;
                h2h[(size_t)row * DIMH + col] = (_Float16)(v > 0.f ? v : 0.f);
            }
        }
    }
}

// ---------------------------------------------------------------------------
// Kernel 7: out = h2h @ w_lin^T + b_lin   (N padded 40 -> 48, 3 N-tiles)
// ---------------------------------------------------------------------------
__global__ __launch_bounds__(256) void k_gemm3(const _Float16* __restrict__ h2h,
                                               const float* __restrict__ wlin,
                                               const float* __restrict__ blin,
                                               float* __restrict__ out) {
    __shared__ __align__(32) _Float16 wlds[NCLS_PAD * DIMH];   // 12 KB, zero-padded
    __shared__ float bl[NCLS_PAD];
    int tid = threadIdx.x;
    for (int i = tid; i < NCLS_PAD * DIMH; i += 256) {
        int n = i >> 7, k = i & 127;
        wlds[i] = (n < NCLS) ? (_Float16)wlin[n * DIMH + k] : (_Float16)0.0f;
    }
    if (tid < NCLS_PAD) bl[tid] = (tid < NCLS) ? blin[tid] : 0.0f;
    __syncthreads();

    int wave = tid >> 5, lane = tid & 31;
    int half = lane >> 4, ln = lane & 15;
    int rowbase = blockIdx.x * 128 + wave * 16;

    v8f acc[3] = {};
    int r = rowbase + ln;

    for (int kt = 0; kt < 4; ++kt) {
        v16h a = {};
        if (r < N_NODES) {
            const _Float16* pa = h2h + (size_t)r * DIMH + kt * 32 + 16 * half;
            v8h lo = *(const v8h*)(pa);
            v8h hi = *(const v8h*)(pa + 8);
            #pragma unroll
            for (int i = 0; i < 8; ++i) { a[i] = lo[i]; a[8 + i] = hi[i]; }
        }
        #pragma unroll
        for (int nt = 0; nt < 3; ++nt) {
            v16h b = *(const v16h*)(wlds + (nt * 16 + ln) * DIMH + kt * 32 + 16 * half);
            acc[nt] = __builtin_amdgcn_wmma_f32_16x16x32_f16(
                false, a, false, b, (short)0, acc[nt], false, false);
        }
    }

    #pragma unroll
    for (int nt = 0; nt < 3; ++nt) {
        int col = nt * 16 + ln;
        if (col >= NCLS) continue;
        float bias = bl[col];
        #pragma unroll
        for (int j = 0; j < 8; ++j) {
            int row = rowbase + j + 8 * half;
            if (row < N_NODES)
                out[(size_t)row * NCLS + col] = acc[nt][j] + bias;
        }
    }
}

// ---------------------------------------------------------------------------
extern "C" void kernel_launch(void* const* d_in, const int* in_sizes, int n_in,
                              void* d_out, int out_size, void* d_ws, size_t ws_size,
                              hipStream_t stream) {
    const float*     x     = (const float*)d_in[0];
    const long long* ei    = (const long long*)d_in[1];
    const float*     w1    = (const float*)d_in[2];
    const float*     b1    = (const float*)d_in[3];
    const float*     gamma = (const float*)d_in[4];
    const float*     beta  = (const float*)d_in[5];
    const float*     w2    = (const float*)d_in[6];
    const float*     b2    = (const float*)d_in[7];
    const float*     wlin  = (const float*)d_in[8];
    const float*     blin  = (const float*)d_in[9];
    float* out = (float*)d_out;

    char* ws = (char*)d_ws;
    float*    h0    = (float*)ws;                         // region A: h0 (f32)
    float*    h1    = (float*)(ws + REGBYTES);            // region B: h1 (f32)
    _Float16* h1h   = (_Float16*)ws;                      // region A reused (f16)
    _Float16* h2h   = (_Float16*)(ws + REGBYTES);         // region B reused (f16)
    float*    stats = (float*)(ws + 2 * REGBYTES);        // 512 floats
    float* gsum  = stats;
    float* gsq   = stats + 128;
    float* scale = stats + 256;
    float* shift = stats + 384;

    const int copy_blocks  = (int)(((size_t)N_NODES * DFEAT / 4 + 255) / 256);  // 12500
    const int edge_blocks  = (int)(((size_t)N_EDGES * 32 + 255) / 256);         // 75000
    const int gemm_blocks  = (N_NODES + 127) / 128;                             // 782

    k_init   <<<copy_blocks, 256, 0, stream>>>(x, h0, stats);
    k_scatter<<<edge_blocks, 256, 0, stream>>>(x, ei, h0);
    k_gemm1  <<<gemm_blocks, 256, 0, stream>>>(h0, w1, b1, h1, gsum, gsq);
    k_bnfin  <<<1, 128, 0, stream>>>(gsum, gsq, gamma, beta, scale, shift);
    k_bncast <<<copy_blocks, 256, 0, stream>>>(h1, scale, shift, h1h);
    k_gemm2  <<<gemm_blocks, 256, 0, stream>>>(h1h, w2, b2, h2h);
    k_gemm3  <<<gemm_blocks, 256, 0, stream>>>(h2h, wlin, blin, out);
}
